// GATv2Classification_78314433675486
// MI455X (gfx1250) — compile-verified, matrix operand order checked by reading, not run on hardware
//
#include <hip/hip_runtime.h>
#include <hip/hip_bf16.h>
#include <stdint.h>

#define N_NODES 50000
#define N_EDGES 800000
#define F_NODE  64
#define F_EDGE  16
#define N_HEAD  2
#define CH      64
#define FDIM    128
#define NLAYER  2
#define NCLS    10
#define LN_EPS  1e-5f
#define SLOPE   0.2f
#define GEMM_ROWS 64

typedef __attribute__((ext_vector_type(16))) __bf16 v16bf;
typedef __attribute__((ext_vector_type(8)))  float  v8f;

union Frag { v16bf v; uint4 q[2]; };
union AccU { v8f v; float f[8]; };
union PK   { __bf16 h[2]; unsigned u; };

__device__ __forceinline__ int flipkey(float f) {
  int i = __float_as_int(f);
  return i >= 0 ? i : (i ^ 0x7fffffff);
}
__device__ __forceinline__ float unflipkey(int k) {
  return __int_as_float(k >= 0 ? k : (k ^ 0x7fffffff));
}

// ---------------------------------------------------------------------------
// Tiled GEMM with bf16 WMMA: out[r][n] = sum_k in[r][k] * W[k][n] + bias[n]
// Fin in {64,128}; FoutPad = Fout rounded up to 16, <= 128.
// Block = 256 threads (8 waves); block tile = 64 rows x FoutPad cols.
// Wave w owns col-tile w for 4 row-tiles; B fragment loaded once per K-step
// and reused across 4 WMMAs (2.5 ds_load_b128 per v_wmma).
// ---------------------------------------------------------------------------
__global__ __launch_bounds__(256)
void k_gemm_bf16(const float* __restrict__ in, const float* __restrict__ W,
                 const float* __restrict__ bias, float* __restrict__ out,
                 int nrows, int Fin, int Fout, int FoutPad)
{
  __shared__ uint4 w4[128 * 17];        // W^T bf16: row n, 128 k-values + pad
  __shared__ uint4 a4[GEMM_ROWS * 17];  // A tile bf16: row r, 128 k-values + pad
  unsigned* wU = reinterpret_cast<unsigned*>(w4);
  unsigned* aU = reinterpret_cast<unsigned*>(a4);

  const int tid = threadIdx.x;
  const int rowBase = blockIdx.x * GEMM_ROWS;
  const int kp2 = Fin >> 1;  // bf16 pairs per row

  // Stage W^T as bf16 pairs: wU[n*68 + kp] holds W[2kp][n], W[2kp+1][n]
  for (int idx = tid; idx < FoutPad * kp2; idx += 256) {
    int n  = idx / kp2;
    int kp = idx - n * kp2;
    PK p; p.u = 0;
    if (n < Fout) {
      p.h[0] = (__bf16)W[(size_t)(2 * kp)     * Fout + n];
      p.h[1] = (__bf16)W[(size_t)(2 * kp + 1) * Fout + n];
    }
    wU[n * 68 + kp] = p.u;
  }
  // Stage A tile as bf16 pairs (float2 global loads): aU[r*68 + kp]
  const float2* in2 = reinterpret_cast<const float2*>(in);
  for (int idx = tid; idx < GEMM_ROWS * kp2; idx += 256) {
    int r  = idx / kp2;
    int kp = idx - r * kp2;
    PK p; p.u = 0;
    int gr = rowBase + r;
    if (gr < nrows) {
      float2 f = in2[(size_t)gr * kp2 + kp];
      p.h[0] = (__bf16)f.x;
      p.h[1] = (__bf16)f.y;
    }
    aU[r * 68 + kp] = p.u;
  }
  __syncthreads();

  const int wv = tid >> 5;
  const int n0 = wv * 16;
  if (n0 >= FoutPad) return;
  const int lane = tid & 31;
  const int hl = lane & 15;
  const int hi = lane >> 4;

  AccU acc[4];
  for (int rt = 0; rt < 4; ++rt)
    for (int i = 0; i < 8; ++i) acc[rt].f[i] = 0.f;

  // ISA 16-bit A layout: lane row = lane&15; K = hi*8 + e (+8 for e>=8),
  // i.e. two contiguous 8-element (16B) runs -> two ds_load_b128 each.
  const int nkstep = Fin >> 5;
  for (int ks = 0; ks < nkstep; ++ks) {
    const int qi = ks * 4 + hi;
    Frag b;
    b.q[0] = w4[(n0 + hl) * 17 + qi];
    b.q[1] = w4[(n0 + hl) * 17 + qi + 2];
#pragma unroll
    for (int rt = 0; rt < 4; ++rt) {
      Frag a;
      a.q[0] = a4[(rt * 16 + hl) * 17 + qi];
      a.q[1] = a4[(rt * 16 + hl) * 17 + qi + 2];
      acc[rt].v = __builtin_amdgcn_wmma_f32_16x16x32_bf16(
          false, a.v, false, b.v, (short)0, acc[rt].v, false, false);
    }
  }

  // D layout: lanes 0-15 hold M=0..7 in VGPR 0..7, lanes 16-31 hold M=8..15.
  const int nn = n0 + hl;
  if (nn < Fout) {
    const float bv = bias[nn];
    for (int rt = 0; rt < 4; ++rt) {
      for (int i = 0; i < 8; ++i) {
        int rr = rowBase + rt * 16 + hi * 8 + i;
        if (rr < nrows) out[(size_t)rr * Fout + nn] = acc[rt].f[i] + bv;
      }
    }
  }
}

// ---------------------------------------------------------------------------
// LayerNorm over F=128: one wave per row, 4 channels/lane, shfl reductions.
// ---------------------------------------------------------------------------
__global__ __launch_bounds__(256)
void k_layernorm(const float* __restrict__ x, const float* __restrict__ g,
                 const float* __restrict__ b, float* __restrict__ y, int nrows)
{
  const int wv = threadIdx.x >> 5, lane = threadIdx.x & 31;
  const int row = blockIdx.x * 8 + wv;
  if (row >= nrows) return;
  float4 v = reinterpret_cast<const float4*>(x + (size_t)row * FDIM)[lane];
  float s = v.x + v.y + v.z + v.w;
  for (int o = 16; o >= 1; o >>= 1) s += __shfl_xor(s, o, 32);
  const float mu = s * (1.f / FDIM);
  float4 d = make_float4(v.x - mu, v.y - mu, v.z - mu, v.w - mu);
  float q = d.x * d.x + d.y * d.y + d.z * d.z + d.w * d.w;
  for (int o = 16; o >= 1; o >>= 1) q += __shfl_xor(q, o, 32);
  const float rs = rsqrtf(q * (1.f / FDIM) + LN_EPS);
  float4 gg = reinterpret_cast<const float4*>(g)[lane];
  float4 bb = reinterpret_cast<const float4*>(b)[lane];
  float4 o4 = make_float4(d.x * rs * gg.x + bb.x, d.y * rs * gg.y + bb.y,
                          d.z * rs * gg.z + bb.z, d.w * rs * gg.w + bb.w);
  reinterpret_cast<float4*>(y + (size_t)row * FDIM)[lane] = o4;
}

// ---------------------------------------------------------------------------
// Per-layer scratch init: segment-max keys -> INT_MIN, den/agg -> 0.
// ---------------------------------------------------------------------------
__global__ void k_init(int* __restrict__ mxkey, float* __restrict__ den,
                       float* __restrict__ agg, int n2, int nagg)
{
  const int t = blockIdx.x * blockDim.x + threadIdx.x;
  if (t < nagg) agg[t] = 0.f;
  if (t < n2) { mxkey[t] = (int)0x80000000u; den[t] = 0.f; }
}

// ---------------------------------------------------------------------------
// Edge logits: grid-stride wave loop, one wave per edge, 4 channels/lane
// (head = lane>>4). ee recomputed from LDS-staged We (never materialized).
// Lookahead global_prefetch_b8 hides the L2 gather latency of the next edge.
// Segment max via atomicMax on monotonic int keys (L2 atomics).
// ---------------------------------------------------------------------------
__global__ __launch_bounds__(256)
void k_edge_logits(const float* __restrict__ xl, const float* __restrict__ xr,
                   const int* __restrict__ ei, const float* __restrict__ eattr,
                   const float* __restrict__ We, const float* __restrict__ att,
                   float* __restrict__ logits, int* __restrict__ mxkey)
{
  __shared__ alignas(16) float we_s[F_EDGE * FDIM];
  __shared__ alignas(16) float att_s[FDIM];
  const int tid = threadIdx.x;
  for (int i = tid; i < F_EDGE * FDIM; i += 256) we_s[i] = We[i];
  if (tid < FDIM) att_s[tid] = att[tid];
  __syncthreads();

  const int lane = tid & 31;
  const int wid = blockIdx.x * 8 + (tid >> 5);
  const int nw  = gridDim.x * 8;
  const int c0  = lane * 4;
  const float4 at = reinterpret_cast<const float4*>(att_s)[lane];

  for (int e = wid; e < N_EDGES; e += nw) {
    const int s = ei[e], d = ei[N_EDGES + e];
    const int en = e + nw;
    if (en < N_EDGES) {
      __builtin_prefetch(xl + (size_t)ei[en] * FDIM + c0, 0, 3);
      __builtin_prefetch(xr + (size_t)ei[N_EDGES + en] * FDIM + c0, 0, 3);
    }
    float4 a = reinterpret_cast<const float4*>(xl + (size_t)s * FDIM)[lane];
    float4 r = reinterpret_cast<const float4*>(xr + (size_t)d * FDIM)[lane];
    float4 ee = make_float4(0.f, 0.f, 0.f, 0.f);
    const float* ea = eattr + (size_t)e * F_EDGE;
    for (int j = 0; j < F_EDGE; ++j) {
      const float w = ea[j];  // same addr across wave -> one request
      const float4 wj = reinterpret_cast<const float4*>(we_s + j * FDIM)[lane];
      ee.x += w * wj.x; ee.y += w * wj.y; ee.z += w * wj.z; ee.w += w * wj.w;
    }
    float4 m = make_float4(a.x + r.x + ee.x, a.y + r.y + ee.y,
                           a.z + r.z + ee.z, a.w + r.w + ee.w);
    m.x = m.x > 0.f ? m.x : SLOPE * m.x;
    m.y = m.y > 0.f ? m.y : SLOPE * m.y;
    m.z = m.z > 0.f ? m.z : SLOPE * m.z;
    m.w = m.w > 0.f ? m.w : SLOPE * m.w;
    float sres = m.x * at.x + m.y * at.y + m.z * at.z + m.w * at.w;
    for (int o = 1; o < 16; o <<= 1) sres += __shfl_xor(sres, o, 32);
    if ((lane & 15) == 0) {
      const int h = lane >> 4;
      logits[(size_t)e * 2 + h] = sres;
      atomicMax(mxkey + (size_t)d * 2 + h, flipkey(sres));
    }
  }
}

// ---------------------------------------------------------------------------
// exp(logit - segmax) + segment-sum of denominators. Thread per (edge, head).
// ---------------------------------------------------------------------------
__global__ void k_edge_exp(const float* __restrict__ logits,
                           const int* __restrict__ mxkey,
                           const int* __restrict__ ei,
                           float* __restrict__ exb, float* __restrict__ den)
{
  const int t = blockIdx.x * blockDim.x + threadIdx.x;
  if (t >= N_EDGES * 2) return;
  const int e = t >> 1, h = t & 1;
  const int d = ei[N_EDGES + e];
  const float mx = unflipkey(mxkey[(size_t)d * 2 + h]);
  const float ex = __expf(logits[t] - mx);
  exb[t] = ex;
  atomicAdd(den + (size_t)d * 2 + h, ex);
}

// ---------------------------------------------------------------------------
// Weighted aggregate: grid-stride wave loop with lookahead prefetch;
// scatter-add xl[src]*alpha into agg via L2 f32 atomics.
// ---------------------------------------------------------------------------
__global__ __launch_bounds__(256)
void k_edge_agg(const float* __restrict__ xl, const int* __restrict__ ei,
                const float* __restrict__ exb, const float* __restrict__ den,
                float* __restrict__ agg)
{
  const int lane = threadIdx.x & 31;
  const int wid = blockIdx.x * 8 + (threadIdx.x >> 5);
  const int nw  = gridDim.x * 8;
  const int c0  = lane * 4;
  const int h   = lane >> 4;

  for (int e = wid; e < N_EDGES; e += nw) {
    const int s = ei[e], d = ei[N_EDGES + e];
    const int en = e + nw;
    if (en < N_EDGES)
      __builtin_prefetch(xl + (size_t)ei[en] * FDIM + c0, 0, 3);
    const float alpha = exb[(size_t)e * 2 + h] / den[(size_t)d * 2 + h];
    const float4 a = reinterpret_cast<const float4*>(xl + (size_t)s * FDIM)[lane];
    float* ap = agg + (size_t)d * FDIM + c0;
    atomicAdd(ap + 0, a.x * alpha);
    atomicAdd(ap + 1, a.y * alpha);
    atomicAdd(ap + 2, a.z * alpha);
    atomicAdd(ap + 3, a.w * alpha);
  }
}

// ---------------------------------------------------------------------------
// h = relu(agg + bout) + h   (residual, in place)
// ---------------------------------------------------------------------------
__global__ void k_node_update(float* __restrict__ h, const float* __restrict__ agg,
                              const float* __restrict__ bout)
{
  const int t = blockIdx.x * blockDim.x + threadIdx.x;
  if (t >= N_NODES * FDIM) return;
  const int c = t & (FDIM - 1);
  const float v = agg[t] + bout[c];
  h[t] = fmaxf(v, 0.f) + h[t];
}

__global__ void k_copy(const float* __restrict__ src, float* __restrict__ dst, int n)
{
  const int t = blockIdx.x * blockDim.x + threadIdx.x;
  if (t < n) dst[t] = src[t];
}

// ---------------------------------------------------------------------------
extern "C" void kernel_launch(void* const* d_in, const int* in_sizes, int n_in,
                              void* d_out, int out_size, void* d_ws, size_t ws_size,
                              hipStream_t stream)
{
  const float* x     = (const float*)d_in[0];
  const int*   ei    = (const int*)  d_in[1];
  const float* eattr = (const float*)d_in[2];
  const float* Wp    = (const float*)d_in[3];
  const float* bp    = (const float*)d_in[4];
  const float* ln_g  = (const float*)d_in[5];
  const float* ln_b  = (const float*)d_in[6];
  const float* Wl    = (const float*)d_in[7];
  const float* bl    = (const float*)d_in[8];
  const float* Wr    = (const float*)d_in[9];
  const float* br    = (const float*)d_in[10];
  const float* We    = (const float*)d_in[11];
  const float* att   = (const float*)d_in[12];
  const float* bout  = (const float*)d_in[13];
  const float* Wc    = (const float*)d_in[14];
  const float* bc    = (const float*)d_in[15];
  float* out = (float*)d_out;

  const size_t NF = (size_t)N_NODES * FDIM;
  float* h      = (float*)d_ws;
  float* hn     = h + NF;
  float* xl     = hn + NF;
  float* xr     = xl + NF;
  float* agg    = xr + NF;
  float* logits = agg + NF;
  float* exb    = logits + (size_t)N_EDGES * 2;
  float* den    = exb + (size_t)N_EDGES * 2;
  int*   mxkey  = (int*)(den + (size_t)N_NODES * 2);

  const dim3 blk(256);
  const int gemmGrid = (N_NODES + GEMM_ROWS - 1) / GEMM_ROWS;
  const int edgeGrid = 2048;  // grid-stride wave loops (~49 edges/wave)
  const int lnGrid   = (N_NODES + 7) / 8;
  const int elemGrid = (int)((NF + 255) / 256);
  const int ehGrid   = (N_EDGES * 2 + 255) / 256;

  // h = x @ Wp + bp
  k_gemm_bf16<<<gemmGrid, blk, 0, stream>>>(x, Wp, bp, h, N_NODES, F_NODE, FDIM, FDIM);

  for (int i = 0; i < NLAYER; ++i) {
    k_layernorm<<<lnGrid, blk, 0, stream>>>(h, ln_g + i * FDIM, ln_b + i * FDIM, hn, N_NODES);
    k_gemm_bf16<<<gemmGrid, blk, 0, stream>>>(hn, Wl + (size_t)i * FDIM * FDIM,
                                              bl + i * FDIM, xl, N_NODES, FDIM, FDIM, FDIM);
    k_gemm_bf16<<<gemmGrid, blk, 0, stream>>>(hn, Wr + (size_t)i * FDIM * FDIM,
                                              br + i * FDIM, xr, N_NODES, FDIM, FDIM, FDIM);
    k_init<<<elemGrid, blk, 0, stream>>>(mxkey, den, agg, N_NODES * 2, (int)NF);
    k_edge_logits<<<edgeGrid, blk, 0, stream>>>(xl, xr, ei, eattr,
                                                We + (size_t)i * F_EDGE * FDIM,
                                                att + i * N_HEAD * CH, logits, mxkey);
    k_edge_exp<<<ehGrid, blk, 0, stream>>>(logits, mxkey, ei, exb, den);
    k_edge_agg<<<edgeGrid, blk, 0, stream>>>(xl, ei, exb, den, agg);
    k_node_update<<<elemGrid, blk, 0, stream>>>(h, agg, bout + i * FDIM);
  }

  // classification_output = h @ Wc + bc  (written first in d_out)
  k_gemm_bf16<<<gemmGrid, blk, 0, stream>>>(h, Wc, bc, out, N_NODES, FDIM, NCLS, 16);
  // node_embeddings = h  (second output, concatenated after [N, K])
  k_copy<<<elemGrid, blk, 0, stream>>>(h, out + (size_t)N_NODES * NCLS, (int)NF);
}
